// DIEN_55482387530397
// MI455X (gfx1250) — matrix operand order repaired and checked by reading, not compiled, not run on hardware
//
#include <hip/hip_runtime.h>
#include <hip/hip_bf16.h>
#include <math.h>

// ---------------- types ----------------
typedef _Float16 h16;
typedef _Float16 v16h __attribute__((ext_vector_type(16)));
typedef _Float16 h8v  __attribute__((ext_vector_type(8)));
typedef float    v8f  __attribute__((ext_vector_type(8)));

#define B_    4096
#define T_    50
#define D_    128
#define DYN_  128
#define REC_V 10000
#define USR_V 10000
#define CTX_V 1000
#define BT_   (B_ * T_)
#define EPS_  1e-5f

// ---------------- WMMA helper (CDNA5 gfx1250) ----------------
__device__ __forceinline__ v8f wmma16(v16h a, v16h b, v8f c) {
  // v_wmma_f32_16x16x32_f16: D = A(16x32 f16) x B(32x16 f16) + C(16x16 f32)
  return __builtin_amdgcn_wmma_f32_16x16x32_f16(false, a, false, b, (short)0, c, false, false);
}
// Opaque zero (SGPR): emits no instruction, but adding it to a load index
// defeats LICM so weight-fragment loads stay inside the time loop (no scratch
// spills) WITHOUT destroying pointer provenance (loads remain global_load,
// not flat_load, so they don't touch DScnt / the LDS pipe).
__device__ __forceinline__ size_t lz() {
  unsigned long long v = 0;
  asm volatile("" : "+s"(v));
  return (size_t)v;
}
// pre-packed B fragment: 512 halfs per (kchunk,ntile), 16 contiguous per lane
__device__ __forceinline__ v16h ldfrag(const h16* __restrict__ base, int NT, int kc, int nt,
                                       int lane, size_t z) {
  return ((const v16h*)base)[(size_t)(kc * NT + nt) * 32 + lane + z];
}
__device__ __forceinline__ float sigf(float x) { return 1.f / (1.f + __expf(-x)); }
__device__ __forceinline__ float pre(float x, float a) { return x >= 0.f ? x : a * x; }

// ---------------- weight -> B-fragment packing ----------------
// B-matrix 32x16 layout (wave32): lane l holds col n = l&15; element e: k = (e<8 ? (l>>4)*8 + e : 16 + (l>>4)*8 + e-8)
__global__ void pack_frag_kernel(const float* __restrict__ W, h16* __restrict__ dst,
                                 int K, int N, int transposed) {
  int idx = blockIdx.x * blockDim.x + threadIdx.x;
  int total = K * N;
  if (idx >= total) return;
  int frag = idx >> 9, w = idx & 511;
  int lane = w >> 4, e = w & 15;
  int NT = N >> 4;
  int kc = frag / NT, nt = frag % NT;
  int k = kc * 32 + (e < 8 ? (lane >> 4) * 8 + e : 16 + (lane >> 4) * 8 + (e - 8));
  int n = nt * 16 + (lane & 15);
  float v = transposed ? W[(size_t)n * K + k] : W[(size_t)k * N + n];
  dst[idx] = (h16)v;
}

// ---------------- embedding gather ----------------
__global__ void gather_kernel(const int* __restrict__ candf, const int* __restrict__ recf,
                              const int* __restrict__ usrf, const int* __restrict__ ctxf,
                              const float* __restrict__ cand_tbl, const float* __restrict__ rec_tbl,
                              const float* __restrict__ usr_tbl, const float* __restrict__ ctx_tbl,
                              float* __restrict__ cand_out, h16* __restrict__ recent_out,
                              h16* __restrict__ dnn_in) {
  int b = blockIdx.x, d = threadIdx.x;  // 128 threads
  int ci = candf[b];
  float cv = cand_tbl[(size_t)ci * D_ + d];
  cand_out[(size_t)b * D_ + d] = cv;
  h16* drow = dnn_in + (size_t)b * 1024;
  drow[d] = (h16)cv;  // cols 0..127: cand
#pragma unroll
  for (int j = 0; j < 4; ++j) {  // cols 256..767: user_flat
    int ui = usrf[b * 4 + j];
    drow[256 + j * D_ + d] = (h16)usr_tbl[((size_t)j * USR_V + ui) * D_ + d];
  }
#pragma unroll
  for (int j = 0; j < 2; ++j) {  // cols 768..1023: ctx_flat
    int xi = ctxf[b * 2 + j];
    drow[768 + j * D_ + d] = (h16)ctx_tbl[((size_t)j * CTX_V + xi) * D_ + d];
  }
  for (int t = 0; t < T_; ++t) {  // per-timestep tables
    int ri = recf[b * T_ + t];
    recent_out[((size_t)b * T_ + t) * D_ + d] = (h16)rec_tbl[((size_t)t * REC_V + ri) * D_ + d];
  }
}

// ---------------- GRU recurrence (batch tile of 32 rows per WG, 4 waves) ----------------
// gates: r=sig(i_r+h_r) z=sig(i_z+h_z) n=tanh(i_n + r*h_n), h'=(1-z)n + z h
__global__ void gru_kernel(const h16* __restrict__ recent, const h16* __restrict__ wx_frag,
                           const h16* __restrict__ wh_frag, const float* __restrict__ b_ih,
                           const float* __restrict__ b_hh, float* __restrict__ Hs) {
  const int rows0 = blockIdx.x * 32;
  const int wave = threadIdx.x >> 5, lane = threadIdx.x & 31;
  const int mtile = wave >> 1;         // 0..1
  const int nc0 = (wave & 1) * 4;      // ncol base within 8
  const int mrow = lane & 15, hi8 = (lane >> 4) * 8;
  const int ccol = lane & 15, cr8 = (lane >> 4) * 8;

  __shared__ float h_lds[32][128];
  for (int i = threadIdx.x; i < 32 * 128; i += blockDim.x) ((float*)h_lds)[i] = 0.f;
  __syncthreads();

  const int rl = mtile * 16 + mrow;
  const int grow = rows0 + rl;
  for (int t = 0; t < T_; ++t) {
    // ---- A fragments: x_t (global f16), h (LDS f32) ----
    v16h ax[4], ah[4];
    const h16* xrow = recent + ((size_t)grow * T_ + t) * D_;
#pragma unroll
    for (int kc = 0; kc < 4; ++kc) {
      const h16* p = xrow + kc * 32 + hi8;
      h8v lo = *(const h8v*)p;
      h8v hv = *(const h8v*)(p + 16);
      const float* hp = &h_lds[rl][kc * 32 + hi8];
#pragma unroll
      for (int i = 0; i < 8; ++i) {
        ax[kc][i] = lo[i];  ax[kc][8 + i] = hv[i];
        ah[kc][i] = (h16)hp[i];  ah[kc][8 + i] = (h16)hp[16 + i];
      }
    }
    __syncthreads();  // all h reads done before any h writes below
#pragma unroll
    for (int c = 0; c < 4; ++c) {
      int nc = nc0 + c;  // 0..7 (16-col chunk within 128)
      v8f accr, accz, accin, acchn;
#pragma unroll
      for (int i = 0; i < 8; ++i) { accr[i] = 0.f; accz[i] = 0.f; accin[i] = 0.f; acchn[i] = 0.f; }
#pragma unroll
      for (int kc = 0; kc < 4; ++kc) {
        size_t zx = lz(), zh = lz();  // per-segment: keeps loads inside t-loop
        accr  = wmma16(ax[kc], ldfrag(wx_frag, 24, kc, nc,      lane, zx), accr);
        accr  = wmma16(ah[kc], ldfrag(wh_frag, 24, kc, nc,      lane, zh), accr);
        accz  = wmma16(ax[kc], ldfrag(wx_frag, 24, kc, 8 + nc,  lane, zx), accz);
        accz  = wmma16(ah[kc], ldfrag(wh_frag, 24, kc, 8 + nc,  lane, zh), accz);
        accin = wmma16(ax[kc], ldfrag(wx_frag, 24, kc, 16 + nc, lane, zx), accin);
        acchn = wmma16(ah[kc], ldfrag(wh_frag, 24, kc, 16 + nc, lane, zh), acchn);
      }
      int col = nc * 16 + ccol;  // 0..127
      float bir = b_ih[col] + b_hh[col];
      float biz = b_ih[128 + col] + b_hh[128 + col];
      float bin = b_ih[256 + col], bhn = b_hh[256 + col];
#pragma unroll
      for (int i = 0; i < 8; ++i) {
        int rr = mtile * 16 + cr8 + i;
        float r = sigf(accr[i] + bir);
        float z = sigf(accz[i] + biz);
        float n = tanhf((accin[i] + bin) + r * (acchn[i] + bhn));
        float hold = h_lds[rr][col];
        float hn = (1.f - z) * n + z * hold;
        h_lds[rr][col] = hn;  // own (rows,cols) region only
        Hs[((size_t)(rows0 + rr) * T_ + t) * D_ + col] = hn;
      }
    }
    __syncthreads();  // h updates visible to next step's fragment loads
  }
}

// ---------------- attention: u=[H,C,H-C,H*C] -> 32 -> 32 -> 1 ----------------
__global__ void att_kernel(const float* __restrict__ Hs, const float* __restrict__ cand,
                           const h16* __restrict__ w1f, const float* __restrict__ b1,
                           const float* __restrict__ a1, const float* __restrict__ w2,
                           const float* __restrict__ b2, const float* __restrict__ a2,
                           const float* __restrict__ w3, const float* __restrict__ b3,
                           float* __restrict__ scores) {
  int wave = threadIdx.x >> 5, lane = threadIdx.x & 31;
  int row0 = (blockIdx.x * 4 + wave) * 16;  // tile base in [0, B*T)
  int m = lane & 15, hi8 = (lane >> 4) * 8;
  int row = row0 + m;
  int b = row / T_;
  const float* hrow = Hs + (size_t)row * D_;
  const float* crow = cand + (size_t)b * D_;
  v8f acc0, acc1;
#pragma unroll
  for (int i = 0; i < 8; ++i) { acc0[i] = 0.f; acc1[i] = 0.f; }
#pragma unroll 1
  for (int kc = 0; kc < 16; ++kc) {
    int kb = kc * 32;
    int sec = kb >> 7;      // 0:H 1:C 2:H-C 3:H*C
    int off = kb & 127;
    v16h a;
#pragma unroll
    for (int i = 0; i < 16; ++i) {
      int k = off + (i < 8 ? hi8 + i : 16 + hi8 + (i - 8));
      float hv = hrow[k], cv = crow[k];
      float v = (sec == 0) ? hv : (sec == 1) ? cv : (sec == 2) ? (hv - cv) : (hv * cv);
      a[i] = (h16)v;
    }
    size_t z = lz();
    acc0 = wmma16(a, ldfrag(w1f, 2, kc, 0, lane, z), acc0);
    acc1 = wmma16(a, ldfrag(w1f, 2, kc, 1, lane, z), acc1);
  }
  __shared__ float x1[4][16][32];
  float a1v = a1[0];
  int ccol = lane & 15, cr8 = (lane >> 4) * 8;
#pragma unroll
  for (int i = 0; i < 8; ++i) {
    x1[wave][cr8 + i][ccol]      = pre(acc0[i] + b1[ccol], a1v);
    x1[wave][cr8 + i][16 + ccol] = pre(acc1[i] + b1[16 + ccol], a1v);
  }
  __syncthreads();
  if (lane < 16) {
    float a2v = a2[0];
    float x2[32];
#pragma unroll 4
    for (int j = 0; j < 32; ++j) {
      float s = b2[j];
      for (int i = 0; i < 32; ++i) s += x1[wave][lane][i] * w2[i * 32 + j];
      x2[j] = pre(s, a2v);
    }
    float s3 = b3[0];
    for (int j = 0; j < 32; ++j) s3 += x2[j] * w3[j];
    scores[row0 + lane] = sigf(s3);
  }
}

// ---------------- AUGRU recurrence (32 rows per WG, 4 waves) ----------------
// r=sig([x,h]Wr+br) u=sig([x,h]Wu+bu) hh=tanh([r*h,x]Wh+bh) u'=a*u h'=(1-u')h+u'hh
__global__ void augru_kernel(const float* __restrict__ Hs, const float* __restrict__ scores,
                             const h16* __restrict__ wr_f, const h16* __restrict__ wu_f,
                             const h16* __restrict__ wh_f, const float* __restrict__ br,
                             const float* __restrict__ bu, const float* __restrict__ bh,
                             h16* __restrict__ dnn_in) {
  const int rows0 = blockIdx.x * 32;
  const int wave = threadIdx.x >> 5, lane = threadIdx.x & 31;
  const int mtile = wave >> 1, nc0 = (wave & 1) * 4;
  const int mrow = lane & 15, hi8 = (lane >> 4) * 8;
  const int ccol = lane & 15, cr8 = (lane >> 4) * 8;

  __shared__ float h_lds[32][128];
  __shared__ float r_lds[32][128];
  for (int i = threadIdx.x; i < 32 * 128; i += blockDim.x) ((float*)h_lds)[i] = 0.f;
  __syncthreads();

  const int rl = mtile * 16 + mrow;
  const int grow = rows0 + rl;
  for (int t = 0; t < T_; ++t) {
    const float* xrow = Hs + ((size_t)grow * T_ + t) * D_;
    // ---- phase 1: A = [x_t, h] (K=256) ----
    v16h af[8];
#pragma unroll
    for (int kc = 0; kc < 8; ++kc) {
      int kb = kc * 32;
#pragma unroll
      for (int i = 0; i < 16; ++i) {
        int k = kb + (i < 8 ? hi8 + i : 16 + hi8 + (i - 8));
        float v = (k < 128) ? xrow[k] : h_lds[rl][k - 128];
        af[kc][i] = (h16)v;
      }
    }
    float ug[4][8];
#pragma unroll
    for (int c = 0; c < 4; ++c) {
      int nc = nc0 + c;
      v8f accr, accu;
#pragma unroll
      for (int i = 0; i < 8; ++i) { accr[i] = 0.f; accu[i] = 0.f; }
#pragma unroll
      for (int kc = 0; kc < 8; ++kc) {
        size_t zr = lz(), zu = lz();
        accr = wmma16(af[kc], ldfrag(wr_f, 8, kc, nc, lane, zr), accr);
        accu = wmma16(af[kc], ldfrag(wu_f, 8, kc, nc, lane, zu), accu);
      }
      int col = nc * 16 + ccol;
      float brc = br[col], buc = bu[col];
#pragma unroll
      for (int i = 0; i < 8; ++i) {
        r_lds[mtile * 16 + cr8 + i][col] = sigf(accr[i] + brc);
        ug[c][i] = sigf(accu[i] + buc);
      }
    }
    __syncthreads();  // r visible to all waves; h still untouched
    // ---- phase 2: A = [r*h, x_t] ----
    v16h bf[8];
#pragma unroll
    for (int kc = 0; kc < 8; ++kc) {
      int kb = kc * 32;
#pragma unroll
      for (int i = 0; i < 16; ++i) {
        int k = kb + (i < 8 ? hi8 + i : 16 + hi8 + (i - 8));
        float v = (k < 128) ? (r_lds[rl][k] * h_lds[rl][k]) : xrow[k - 128];
        bf[kc][i] = (h16)v;
      }
    }
    __syncthreads();  // all fragment reads of h done before updates
#pragma unroll
    for (int c = 0; c < 4; ++c) {
      int nc = nc0 + c;
      v8f acch;
#pragma unroll
      for (int i = 0; i < 8; ++i) acch[i] = 0.f;
#pragma unroll
      for (int kc = 0; kc < 8; ++kc) {
        size_t zh = lz();
        acch = wmma16(bf[kc], ldfrag(wh_f, 8, kc, nc, lane, zh), acch);
      }
      int col = nc * 16 + ccol;
      float bhc = bh[col];
#pragma unroll
      for (int i = 0; i < 8; ++i) {
        int rr = mtile * 16 + cr8 + i;
        float hhat = tanhf(acch[i] + bhc);
        float at = scores[(size_t)(rows0 + rr) * T_ + t];
        float up = at * ug[c][i];
        float hold = h_lds[rr][col];
        float hn = (1.f - up) * hold + up * hhat;
        h_lds[rr][col] = hn;
        if (t == T_ - 1)  // h_fin -> dnn_in cols 128..255
          dnn_in[(size_t)(rows0 + rr) * 1024 + 128 + col] = (h16)hn;
      }
    }
    __syncthreads();
  }
}

// ---------------- generic WMMA GEMM: Y[rows,N] = A[rows,K](f16) @ B + bias ----------------
template <int KC, int NT>
__global__ void gemm_kernel(const h16* __restrict__ A, const h16* __restrict__ Bf,
                            const float* __restrict__ bias, float* __restrict__ Y,
                            int K, int N) {
  int wave = threadIdx.x >> 5, lane = threadIdx.x & 31;
  int row0 = (blockIdx.x * 4 + wave) * 16;
  int m = lane & 15, hi8 = (lane >> 4) * 8;
  const h16* arow = A + (size_t)(row0 + m) * K;
  v8f acc[NT];
#pragma unroll
  for (int nt = 0; nt < NT; ++nt)
#pragma unroll
    for (int i = 0; i < 8; ++i) acc[nt][i] = 0.f;
#pragma unroll 1
  for (int kc = 0; kc < KC; ++kc) {
    const h16* p = arow + kc * 32 + hi8;
    h8v lo = *(const h8v*)p;
    h8v hv = *(const h8v*)(p + 16);
    v16h a;
#pragma unroll
    for (int i = 0; i < 8; ++i) { a[i] = lo[i]; a[8 + i] = hv[i]; }
    size_t z = lz();
#pragma unroll
    for (int nt = 0; nt < NT; ++nt) acc[nt] = wmma16(a, ldfrag(Bf, NT, kc, nt, lane, z), acc[nt]);
  }
  int ccol = lane & 15, cr8 = (lane >> 4) * 8;
#pragma unroll
  for (int nt = 0; nt < NT; ++nt) {
    int col = nt * 16 + ccol;
    float bb = bias[col];
#pragma unroll
    for (int i = 0; i < 8; ++i) Y[(size_t)(row0 + cr8 + i) * N + col] = acc[nt][i] + bb;
  }
}

// ---------------- batch-norm stats (training mode: batch mean, biased var) ----------------
__global__ void bn_stats_kernel(const float* __restrict__ Y, float* __restrict__ stats,
                                int rows, int cols) {
  int col = blockIdx.x, tid = threadIdx.x;
  __shared__ float ss[256], sq[256];
  float s = 0.f, q = 0.f;
  for (int r = tid; r < rows; r += 256) {
    float v = Y[(size_t)r * cols + col];
    s += v; q += v * v;
  }
  ss[tid] = s; sq[tid] = q;
  __syncthreads();
  for (int st = 128; st > 0; st >>= 1) {
    if (tid < st) { ss[tid] += ss[tid + st]; sq[tid] += sq[tid + st]; }
    __syncthreads();
  }
  if (tid == 0) {
    float mn = ss[0] / rows;
    float var = sq[0] / rows - mn * mn;
    stats[col] = mn;
    stats[cols + col] = rsqrtf(var + EPS_);
  }
}

// ---------------- BN + PReLU -> f16 activations ----------------
__global__ void bn_act_kernel(const float* __restrict__ Y, const float* __restrict__ stats,
                              const float* __restrict__ g, const float* __restrict__ bta,
                              const float* __restrict__ a, h16* __restrict__ X,
                              int rows, int cols) {
  int idx = blockIdx.x * blockDim.x + threadIdx.x;
  if (idx >= rows * cols) return;
  int col = idx % cols;
  float v = g[col] * (Y[idx] - stats[col]) * stats[cols + col] + bta[col];
  X[idx] = (h16)pre(v, a[0]);
}

// ---------------- final: BN2 + PReLU + Linear(64->1) + sigmoid ----------------
__global__ void final_kernel(const float* __restrict__ Y2, const float* __restrict__ stats,
                             const float* __restrict__ g, const float* __restrict__ bta,
                             const float* __restrict__ a, const float* __restrict__ w3,
                             const float* __restrict__ b3, float* __restrict__ out) {
  int row = blockIdx.x * blockDim.x + threadIdx.x;
  if (row >= B_) return;
  float av = a[0];
  float s = b3[0];
  for (int j = 0; j < 64; ++j) {
    float v = g[j] * (Y2[(size_t)row * 64 + j] - stats[j]) * stats[64 + j] + bta[j];
    s += pre(v, av) * w3[j];
  }
  out[row] = sigf(s);
}

// ---------------- host launch ----------------
extern "C" void kernel_launch(void* const* d_in, const int* in_sizes, int n_in,
                              void* d_out, int out_size, void* d_ws, size_t ws_size,
                              hipStream_t stream) {
  (void)in_sizes; (void)n_in; (void)out_size; (void)ws_size;
  const int*   candf    = (const int*)d_in[0];
  const int*   recf     = (const int*)d_in[1];
  const int*   usrf     = (const int*)d_in[2];
  const int*   ctxf     = (const int*)d_in[3];
  const float* cand_tbl = (const float*)d_in[4];
  const float* rec_tbl  = (const float*)d_in[5];
  const float* usr_tbl  = (const float*)d_in[6];
  const float* ctx_tbl  = (const float*)d_in[7];
  const float* gru_w_ih = (const float*)d_in[8];
  const float* gru_w_hh = (const float*)d_in[9];
  const float* gru_b_ih = (const float*)d_in[10];
  const float* gru_b_hh = (const float*)d_in[11];
  const float* att_w1   = (const float*)d_in[12];
  const float* att_b1   = (const float*)d_in[13];
  const float* att_a1   = (const float*)d_in[14];
  const float* att_w2   = (const float*)d_in[15];
  const float* att_b2   = (const float*)d_in[16];
  const float* att_a2   = (const float*)d_in[17];
  const float* att_w3   = (const float*)d_in[18];
  const float* att_b3   = (const float*)d_in[19];
  const float* aug_wr   = (const float*)d_in[20];
  const float* aug_br   = (const float*)d_in[21];
  const float* aug_wu   = (const float*)d_in[22];
  const float* aug_bu   = (const float*)d_in[23];
  const float* aug_wh   = (const float*)d_in[24];
  const float* aug_bh   = (const float*)d_in[25];
  const float* dnn_w1   = (const float*)d_in[26];
  const float* dnn_b1   = (const float*)d_in[27];
  const float* bn1_g    = (const float*)d_in[28];
  const float* bn1_b    = (const float*)d_in[29];
  const float* dnn_a1   = (const float*)d_in[30];
  const float* dnn_w2   = (const float*)d_in[31];
  const float* dnn_b2   = (const float*)d_in[32];
  const float* bn2_g    = (const float*)d_in[33];
  const float* bn2_b    = (const float*)d_in[34];
  const float* dnn_a2   = (const float*)d_in[35];
  const float* dnn_w3   = (const float*)d_in[36];
  const float* dnn_b3   = (const float*)d_in[37];
  float* out = (float*)d_out;

  char* ws = (char*)d_ws;
  size_t off = 0;
  auto alloc = [&](size_t bytes) -> size_t {
    size_t r = off;
    off += (bytes + 255) & ~(size_t)255;
    return r;
  };
  size_t o_recent = alloc((size_t)BT_ * D_ * 2);     // f16 history embeddings
  size_t o_H      = alloc((size_t)BT_ * D_ * 4);     // GRU hidden states (f32)
  size_t o_scores = alloc((size_t)BT_ * 4);
  size_t o_cand   = alloc((size_t)B_ * D_ * 4);
  size_t o_dnnin  = alloc((size_t)B_ * 1024 * 2);
  size_t o_y1     = alloc((size_t)B_ * 128 * 4);
  size_t o_y2     = alloc((size_t)B_ * 64 * 4);
  size_t o_x1h    = alloc((size_t)B_ * 128 * 2);
  size_t o_st1    = alloc(256 * 4);
  size_t o_st2    = alloc(128 * 4);
  size_t o_wx     = alloc((size_t)128 * 384 * 2);
  size_t o_whh    = alloc((size_t)128 * 384 * 2);
  size_t o_att1   = alloc((size_t)512 * 32 * 2);
  size_t o_augr   = alloc((size_t)256 * 128 * 2);
  size_t o_augu   = alloc((size_t)256 * 128 * 2);
  size_t o_augh   = alloc((size_t)256 * 128 * 2);
  size_t o_d1     = alloc((size_t)1024 * 128 * 2);
  size_t o_d2     = alloc((size_t)128 * 64 * 2);

  h16*   recent_h = (h16*)(ws + o_recent);
  float* Hbuf     = (float*)(ws + o_H);
  float* scores   = (float*)(ws + o_scores);
  float* cand     = (float*)(ws + o_cand);
  h16*   dnnin    = (h16*)(ws + o_dnnin);
  float* y1       = (float*)(ws + o_y1);
  float* y2       = (float*)(ws + o_y2);
  h16*   x1h      = (h16*)(ws + o_x1h);
  float* st1      = (float*)(ws + o_st1);
  float* st2      = (float*)(ws + o_st2);
  h16*   wx_f     = (h16*)(ws + o_wx);
  h16*   wh_f     = (h16*)(ws + o_whh);
  h16*   att1_f   = (h16*)(ws + o_att1);
  h16*   augr_f   = (h16*)(ws + o_augr);
  h16*   augu_f   = (h16*)(ws + o_augu);
  h16*   augh_f   = (h16*)(ws + o_augh);
  h16*   d1_f     = (h16*)(ws + o_d1);
  h16*   d2_f     = (h16*)(ws + o_d2);

  auto packGrid = [](int K, int N) { return dim3((K * N + 255) / 256); };
  // weight -> WMMA B-fragment packing (transposed=1: src is [N][K] e.g. torch W_ih)
  pack_frag_kernel<<<packGrid(128, 384), 256, 0, stream>>>(gru_w_ih, wx_f, 128, 384, 1);
  pack_frag_kernel<<<packGrid(128, 384), 256, 0, stream>>>(gru_w_hh, wh_f, 128, 384, 1);
  pack_frag_kernel<<<packGrid(512, 32),  256, 0, stream>>>(att_w1,  att1_f, 512, 32, 0);
  pack_frag_kernel<<<packGrid(256, 128), 256, 0, stream>>>(aug_wr,  augr_f, 256, 128, 0);
  pack_frag_kernel<<<packGrid(256, 128), 256, 0, stream>>>(aug_wu,  augu_f, 256, 128, 0);
  pack_frag_kernel<<<packGrid(256, 128), 256, 0, stream>>>(aug_wh,  augh_f, 256, 128, 0);
  pack_frag_kernel<<<packGrid(1024, 128), 256, 0, stream>>>(dnn_w1, d1_f, 1024, 128, 0);
  pack_frag_kernel<<<packGrid(128, 64),  256, 0, stream>>>(dnn_w2,  d2_f, 128, 64, 0);

  // embeddings
  gather_kernel<<<B_, 128, 0, stream>>>(candf, recf, usrf, ctxf, cand_tbl, rec_tbl, usr_tbl,
                                        ctx_tbl, cand, recent_h, dnnin);
  // GRU over T (batch-tiled, fully fused recurrence)
  gru_kernel<<<B_ / 32, 128, 0, stream>>>(recent_h, wx_f, wh_f, gru_b_ih, gru_b_hh, Hbuf);
  // attention scores
  att_kernel<<<BT_ / 64, 128, 0, stream>>>(Hbuf, cand, att1_f, att_b1, att_a1, att_w2, att_b2,
                                           att_a2, att_w3, att_b3, scores);
  // AUGRU over T; writes h_fin into dnn_in cols 128..255
  augru_kernel<<<B_ / 32, 128, 0, stream>>>(Hbuf, scores, augr_f, augu_f, augh_f, aug_br, aug_bu,
                                            aug_bh, dnnin);
  // DNN head with training-mode BatchNorm
  gemm_kernel<32, 8><<<B_ / 64, 128, 0, stream>>>(dnnin, d1_f, dnn_b1, y1, 1024, 128);
  bn_stats_kernel<<<128, 256, 0, stream>>>(y1, st1, B_, 128);
  bn_act_kernel<<<(B_ * 128 + 255) / 256, 256, 0, stream>>>(y1, st1, bn1_g, bn1_b, dnn_a1, x1h,
                                                            B_, 128);
  gemm_kernel<4, 4><<<B_ / 64, 128, 0, stream>>>(x1h, d2_f, dnn_b2, y2, 128, 64);
  bn_stats_kernel<<<64, 256, 0, stream>>>(y2, st2, B_, 64);
  final_kernel<<<(B_ + 255) / 256, 256, 0, stream>>>(y2, st2, bn2_g, bn2_b, dnn_a2, dnn_w3,
                                                     dnn_b3, out);
}
